// ColTrastModel_6820408066807
// MI455X (gfx1250) — matrix operand order repaired and checked by reading, not compiled, not run on hardware
//
#include <hip/hip_runtime.h>

// ---------------------------------------------------------------------------
// ColTrast on MI455X (gfx1250, wave32).
// Dominant cost: per-batch late-interaction GEMM [128x512x384] x 256 batches
// = 12.9 GFLOP -> run on v_wmma_f32_16x16x32_bf16 after per-token L2 norm,
// so the whole pipeline is HBM-bound (~500MB @ 23.3 TB/s).
// ---------------------------------------------------------------------------

typedef __attribute__((ext_vector_type(16))) __bf16 v16bf;
typedef __attribute__((ext_vector_type(8)))  float  v8f;

#define DEV static __device__ __forceinline__

static constexpr int   Dm      = 384;
static constexpr int   LQ      = 128;
static constexpr int   LC      = 512;
static constexpr int   NB      = 256;   // batch
static constexpr int   AST     = 392;   // padded LDS row stride (bf16 elems): 784B -> 4-bank skew
static constexpr float NEG_INF = -1e9f;

DEV unsigned short f2bf(float f) {
  unsigned int u = __float_as_uint(f);
  u += 0x7FFFu + ((u >> 16) & 1u);      // round-to-nearest-even
  return (unsigned short)(u >> 16);
}

// A-fragment (16x32 bf16, MxK): lane half 0 -> K {0..7} & {16..23}; half 1 -> +8.
// p points at row*AST + k0 + (lanehalf*8); reads 8 elems at +0 and 8 at +16.
DEV v16bf ld_frag_a(const unsigned short* p) {
  v16bf f;
  uint4* fp = reinterpret_cast<uint4*>(&f);
  fp[0] = *reinterpret_cast<const uint4*>(p);
  fp[1] = *reinterpret_cast<const uint4*>(p + 16);
  return f;
}

// B-fragment (32x16 bf16, KxN): lane holds column N=lane%16; half 0 -> K {0..7} & {16..23};
// half 1 -> +8. p points at n*AST + k0 + (lanehalf*8); reads 8 elems at +0 and 8 at +16.
DEV v16bf ld_frag_b(const unsigned short* p) {
  v16bf f;
  uint4* fp = reinterpret_cast<uint4*>(&f);
  fp[0] = *reinterpret_cast<const uint4*>(p);
  fp[1] = *reinterpret_cast<const uint4*>(p + 16);
  return f;
}

DEV v8f wmma_bf16(v16bf a, v16bf b, v8f c) {
  return __builtin_amdgcn_wmma_f32_16x16x32_bf16(false, a, false, b,
                                                 (short)0, c, false, false);
}

// ---------------------------------------------------------------------------
// Kernel 1: fused masked mean-pool (over L) + per-token inverse L2 norm.
// One block per batch; 8 waves; wave w handles tokens l = w, w+8, ...
// Lane loads 3 coalesced float4 (d = 4*lane + 128*k).
// ---------------------------------------------------------------------------
__global__ __launch_bounds__(256) void ct_pool_norm(
    const float* __restrict__ tok, const int* __restrict__ mask, int L,
    float* __restrict__ pooled, float* __restrict__ inv) {
  __shared__ float pp[8 * Dm];
  __shared__ float cw[8];
  const int b = blockIdx.x, tid = threadIdx.x;
  const int wid = tid >> 5, lane = tid & 31;

  float acc[12];
#pragma unroll
  for (int i = 0; i < 12; ++i) acc[i] = 0.f;
  float cnt = 0.f;

  for (int l = wid; l < L; l += 8) {
    const float* rp = tok + ((size_t)b * L + l) * Dm;
    const int m = mask[b * L + l];
    float x[12];
    float ss = 0.f;
#pragma unroll
    for (int k = 0; k < 3; ++k) {
      const float4 v = *reinterpret_cast<const float4*>(rp + (k << 7) + (lane << 2));
      x[k * 4 + 0] = v.x; x[k * 4 + 1] = v.y; x[k * 4 + 2] = v.z; x[k * 4 + 3] = v.w;
      ss += v.x * v.x + v.y * v.y + v.z * v.z + v.w * v.w;
    }
#pragma unroll
    for (int s = 16; s >= 1; s >>= 1) ss += __shfl_xor(ss, s, 32);
    if (lane == 0) {
      inv[b * L + l] = 1.f / fmaxf(sqrtf(ss), 1e-12f);
      cnt += (float)m;
    }
    if (m) {
#pragma unroll
      for (int i = 0; i < 12; ++i) acc[i] += x[i];
    }
  }

#pragma unroll
  for (int k = 0; k < 3; ++k) {
#pragma unroll
    for (int j = 0; j < 4; ++j)
      pp[wid * Dm + (k << 7) + (lane << 2) + j] = acc[k * 4 + j];
  }
  if (lane == 0) cw[wid] = cnt;
  __syncthreads();

  float ctot = 0.f;
#pragma unroll
  for (int w = 0; w < 8; ++w) ctot += cw[w];
  ctot = fmaxf(ctot, 1e-9f);
  for (int d = tid; d < Dm; d += 256) {
    float s = 0.f;
#pragma unroll
    for (int w = 0; w < 8; ++w) s += pp[w * Dm + d];
    pooled[b * Dm + d] = s / ctot;
  }
}

// ---------------------------------------------------------------------------
// Kernel 2a: h = relu(pooled @ W1 + b1). 512 rows (q then c). Tiny: f32 VALU.
// ---------------------------------------------------------------------------
__global__ __launch_bounds__(128) void ct_mlp1(
    const float* __restrict__ qp, const float* __restrict__ cp,
    const float* __restrict__ W1, const float* __restrict__ b1,
    float* __restrict__ h) {
  __shared__ float xs[Dm];
  const int r = blockIdx.x, tid = threadIdx.x;
  const float* x = (r < NB) ? (qp + r * Dm) : (cp + (r - NB) * Dm);
  for (int i = tid; i < Dm; i += 128) xs[i] = x[i];
  __syncthreads();
  for (int j = tid; j < Dm; j += 128) {
    float a = b1[j];
    for (int k = 0; k < Dm; ++k) a = fmaf(xs[k], W1[k * Dm + j], a);
    h[r * Dm + j] = fmaxf(a, 0.f);
  }
}

// ---------------------------------------------------------------------------
// Kernel 2b: proj = h @ W2 + b2; row L2-norm; emit bf16 embeddings.
// ---------------------------------------------------------------------------
__global__ __launch_bounds__(128) void ct_mlp2(
    const float* __restrict__ h, const float* __restrict__ W2,
    const float* __restrict__ b2, unsigned short* __restrict__ ebf) {
  __shared__ float xs[Dm];
  __shared__ float ps[Dm];
  __shared__ float rs[128];
  const int r = blockIdx.x, tid = threadIdx.x;
  for (int i = tid; i < Dm; i += 128) xs[i] = h[r * Dm + i];
  __syncthreads();
  float ssq = 0.f;
  for (int j = tid; j < Dm; j += 128) {
    float a = b2[j];
    for (int k = 0; k < Dm; ++k) a = fmaf(xs[k], W2[k * Dm + j], a);
    ps[j] = a;
    ssq += a * a;
  }
  rs[tid] = ssq;
  __syncthreads();
  for (int s = 64; s > 0; s >>= 1) {
    if (tid < s) rs[tid] += rs[tid + s];
    __syncthreads();
  }
  const float invn = 1.f / fmaxf(sqrtf(rs[0]), 1e-12f);
  for (int j = tid; j < Dm; j += 128) ebf[r * Dm + j] = f2bf(ps[j] * invn);
}

// ---------------------------------------------------------------------------
// Kernel 3: late interaction. One block per batch (256 thr = 8 waves).
// LDS: A = normalized q tokens 128x384 bf16 (stride 392), B = 64-row c chunk.
// Wave w owns M-tile w; 12 resident A-fragments; 8 chunks x 4 N-tiles x 12 WMMA.
// Masked max over N in C-register layout, xor-shuffle reduce, q-mask, sum.
// ---------------------------------------------------------------------------
__global__ __launch_bounds__(256) void ct_late(
    const float* __restrict__ qtok, const float* __restrict__ ctok,
    const int* __restrict__ qmask, const int* __restrict__ cmask,
    const float* __restrict__ qinv, const float* __restrict__ cinv,
    float* __restrict__ late) {
  extern __shared__ unsigned short sm[];
  unsigned short* As = sm;                 // 128*392 bf16
  unsigned short* Bs = sm + LQ * AST;      // 64*392 bf16
  __shared__ int   smc[64];
  __shared__ float sred[16];

  const int b = blockIdx.x, tid = threadIdx.x;
  const int wid = tid >> 5, lane = tid & 31;
  const int lhalf = lane >> 4, l16 = lane & 15;

  // stage normalized q tokens as bf16
  const float* qb = qtok + (size_t)b * LQ * Dm;
  for (int c = tid; c < LQ * (Dm / 4); c += 256) {
    const int row = c / (Dm / 4), c4 = c - row * (Dm / 4);
    const float iv = qinv[b * LQ + row];
    const float4 v = *reinterpret_cast<const float4*>(qb + row * Dm + c4 * 4);
    ushort4 o;
    o.x = f2bf(v.x * iv); o.y = f2bf(v.y * iv);
    o.z = f2bf(v.z * iv); o.w = f2bf(v.w * iv);
    *reinterpret_cast<ushort4*>(As + row * AST + c4 * 4) = o;
  }
  __syncthreads();

  v16bf afrag[12];
  {
    const unsigned short* ar = As + ((wid << 4) + l16) * AST + (lhalf << 3);
#pragma unroll
    for (int k = 0; k < 12; ++k) afrag[k] = ld_frag_a(ar + 32 * k);
  }

  float mx[8];
#pragma unroll
  for (int r = 0; r < 8; ++r) mx[r] = NEG_INF;

  const float* cb = ctok + (size_t)b * LC * Dm;
  for (int ch = 0; ch < 8; ++ch) {
    __syncthreads();  // previous chunk fully consumed
    for (int c = tid; c < 64 * (Dm / 4); c += 256) {
      const int row = c / (Dm / 4), c4 = c - row * (Dm / 4);
      const int n = (ch << 6) + row;
      const float iv = cinv[b * LC + n];
      const float4 v = *reinterpret_cast<const float4*>(cb + n * Dm + c4 * 4);
      ushort4 o;
      o.x = f2bf(v.x * iv); o.y = f2bf(v.y * iv);
      o.z = f2bf(v.z * iv); o.w = f2bf(v.w * iv);
      *reinterpret_cast<ushort4*>(Bs + row * AST + c4 * 4) = o;
    }
    if (tid < 64) smc[tid] = cmask[b * LC + (ch << 6) + tid];
    __syncthreads();

#pragma unroll
    for (int nt = 0; nt < 4; ++nt) {
      const unsigned short* br = Bs + ((nt << 4) + l16) * AST + (lhalf << 3);
      v8f acc = {0.f, 0.f, 0.f, 0.f, 0.f, 0.f, 0.f, 0.f};
#pragma unroll
      for (int k = 0; k < 12; ++k) {
        const v16bf bf = ld_frag_b(br + 32 * k);
        acc = wmma_bf16(afrag[k], bf, acc);
      }
      const int cm = smc[(nt << 4) + l16];  // column N = lane%16
#pragma unroll
      for (int r = 0; r < 8; ++r) mx[r] = fmaxf(mx[r], cm ? acc[r] : NEG_INF);
    }
  }

  // reduce max over the 16 lanes of each half (N dimension)
#pragma unroll
  for (int r = 0; r < 8; ++r) {
#pragma unroll
    for (int s = 8; s >= 1; s >>= 1) mx[r] = fmaxf(mx[r], __shfl_xor(mx[r], s, 32));
  }
  if (l16 == 0) {
    float s = 0.f;
#pragma unroll
    for (int r = 0; r < 8; ++r) {
      const int qrow = (wid << 4) + (lhalf << 3) + r;  // M = r + 8*lanehalf
      s += qmask[b * LQ + qrow] ? mx[r] : 0.f;
    }
    sred[(wid << 1) + lhalf] = s;
  }
  __syncthreads();
  if (tid == 0) {
    float s = 0.f;
    for (int i = 0; i < 16; ++i) s += sred[i];
    late[b] = s;
  }
}

// ---------------------------------------------------------------------------
// Kernel 4: contrastive = (q_emb @ c_emb^T) / 0.07 via WMMA bf16.
// Grid (2,4): 128-row M block x 64-col N block; 8 waves of M-tiles.
// ---------------------------------------------------------------------------
__global__ __launch_bounds__(256) void ct_contrastive(
    const unsigned short* __restrict__ ebf, float* __restrict__ out) {
  extern __shared__ unsigned short sm[];
  unsigned short* As = sm;             // 128*392
  unsigned short* Bs = sm + 128 * AST; // 64*392
  const int mblk = blockIdx.x, nblk = blockIdx.y, tid = threadIdx.x;
  const int wid = tid >> 5, lane = tid & 31;
  const int lhalf = lane >> 4, l16 = lane & 15;
  const unsigned short* qe = ebf;
  const unsigned short* ce = ebf + NB * Dm;

  for (int c = tid; c < 128 * (Dm / 4); c += 256) {
    const int row = c / (Dm / 4), c4 = c - row * (Dm / 4);
    *reinterpret_cast<ushort4*>(As + row * AST + c4 * 4) =
        *reinterpret_cast<const ushort4*>(qe + (mblk * 128 + row) * Dm + c4 * 4);
  }
  for (int c = tid; c < 64 * (Dm / 4); c += 256) {
    const int row = c / (Dm / 4), c4 = c - row * (Dm / 4);
    *reinterpret_cast<ushort4*>(Bs + row * AST + c4 * 4) =
        *reinterpret_cast<const ushort4*>(ce + (nblk * 64 + row) * Dm + c4 * 4);
  }
  __syncthreads();

  v16bf afrag[12];
  {
    const unsigned short* ar = As + ((wid << 4) + l16) * AST + (lhalf << 3);
#pragma unroll
    for (int k = 0; k < 12; ++k) afrag[k] = ld_frag_a(ar + 32 * k);
  }
  const float sc = 1.0f / 0.07f;
#pragma unroll
  for (int nt = 0; nt < 4; ++nt) {
    const unsigned short* br = Bs + ((nt << 4) + l16) * AST + (lhalf << 3);
    v8f acc = {0.f, 0.f, 0.f, 0.f, 0.f, 0.f, 0.f, 0.f};
#pragma unroll
    for (int k = 0; k < 12; ++k) {
      const v16bf bf = ld_frag_b(br + 32 * k);
      acc = wmma_bf16(afrag[k], bf, acc);
    }
#pragma unroll
    for (int r = 0; r < 8; ++r) {
      const int row = mblk * 128 + (wid << 4) + (lhalf << 3) + r;
      const int col = nblk * 64 + (nt << 4) + l16;
      out[row * NB + col] = acc[r] * sc;
    }
  }
}

// ---------------------------------------------------------------------------
extern "C" void kernel_launch(void* const* d_in, const int* in_sizes, int n_in,
                              void* d_out, int out_size, void* d_ws, size_t ws_size,
                              hipStream_t stream) {
  const float* qtok  = (const float*)d_in[0];
  const float* ctok  = (const float*)d_in[1];
  const int*   qmask = (const int*)d_in[2];
  const int*   cmask = (const int*)d_in[3];
  const float* W1    = (const float*)d_in[4];
  const float* b1    = (const float*)d_in[5];
  const float* W2    = (const float*)d_in[6];
  const float* b2    = (const float*)d_in[7];
  float* out = (float*)d_out;

  // workspace layout (floats)
  float* ws    = (float*)d_ws;
  float* qpool = ws;                       // 256*384
  float* cpool = qpool + NB * Dm;          // 256*384
  float* qinv  = cpool + NB * Dm;          // 256*128
  float* cinv  = qinv + NB * LQ;           // 256*512
  float* hbuf  = cinv + NB * LC;           // 512*384
  unsigned short* ebf = (unsigned short*)(hbuf + 2 * NB * Dm);  // 512*384 bf16

  const size_t ldsBytes = (size_t)(128 * AST + 64 * AST) * sizeof(unsigned short); // 150528
  (void)hipFuncSetAttribute(reinterpret_cast<const void*>(ct_late),
                            hipFuncAttributeMaxDynamicSharedMemorySize, (int)ldsBytes);
  (void)hipFuncSetAttribute(reinterpret_cast<const void*>(ct_contrastive),
                            hipFuncAttributeMaxDynamicSharedMemorySize, (int)ldsBytes);

  ct_pool_norm<<<NB, 256, 0, stream>>>(qtok, qmask, LQ, qpool, qinv);
  ct_pool_norm<<<NB, 256, 0, stream>>>(ctok, cmask, LC, cpool, cinv);
  ct_mlp1<<<2 * NB, 128, 0, stream>>>(qpool, cpool, W1, b1, hbuf);
  ct_mlp2<<<2 * NB, 128, 0, stream>>>(hbuf, W2, b2, ebf);
  ct_contrastive<<<dim3(2, 4), 256, ldsBytes, stream>>>(ebf, out);
  ct_late<<<NB, 256, ldsBytes, stream>>>(qtok, ctok, qmask, cmask, qinv, cinv,
                                         out + NB * NB);
}
